// GNNStack_60524679135855
// MI455X (gfx1250) — matrix operand order repaired and checked by reading, not compiled, not run on hardware
//
#include <hip/hip_runtime.h>
#include <hip/hip_bf16.h>
#include <math.h>

typedef __attribute__((ext_vector_type(16))) _Float16 v16h;
typedef __attribute__((ext_vector_type(8)))  float    v8f;

#define HEADS 2
#define HID   64
#define F     128   // HEADS*HID
#define NEG_SLOPE 0.2f

// ---------------------------------------------------------------------------
// Pack W[K,Ncols] (f32, row-major) into per-wave WMMA B fragments (f16),
// zero-padded to 16-col tiles. Fragment layout so that GEMM lane `l` reads its
// v16h with a single contiguous 32B access:
//   out[ ((tile*Ksteps + kt)*32 + lane)*16 + j ]
//   col = tile*16 + (lane&15);  k = kt*32 + (lane>>4)*16 + j
// ---------------------------------------------------------------------------
__global__ void pack_w(const float* __restrict__ W, _Float16* __restrict__ out,
                       int K, int Ncols, int colTiles)
{
    int idx   = blockIdx.x * blockDim.x + threadIdx.x;
    int Ksteps = K >> 5;
    int total = colTiles * Ksteps * 32 * 16;
    if (idx >= total) return;
    int j    = idx & 15;
    int lane = (idx >> 4) & 31;
    int kf   = idx >> 9;                 // tile*Ksteps + kt
    int kt   = kf % Ksteps;
    int tile = kf / Ksteps;
    int col  = tile * 16 + (lane & 15);
    int k    = kt * 32 + (lane >> 4) * 16 + j;
    out[idx] = (col < Ncols) ? (_Float16)W[(size_t)k * Ncols + col] : (_Float16)0.f;
}

// ---------------------------------------------------------------------------
// WMMA GEMM: C[Nrows,Ncols] = A[Nrows,K] @ W[K,Ncols] + bias, optional ReLU.
// One wave computes a 16x(16*NT) strip: A fragment loaded once per K-step,
// NT back-to-back v_wmma_f32_16x16x32_f16 with independent accumulators.
// Requires colTiles % NT == 0 (B zero-padded at pack time).
// ---------------------------------------------------------------------------
template<int NT>
__global__ void gemm_wmma_packed(const float* __restrict__ A, const _Float16* __restrict__ Wpk,
                                 const float* __restrict__ bias, float* __restrict__ C,
                                 int Nrows, int K, int Ncols, int doRelu)
{
    const int lane      = threadIdx.x & 31;
    const int waveId    = blockIdx.x * (blockDim.x >> 5) + (threadIdx.x >> 5);
    const int colTiles  = (Ncols + 15) >> 4;
    const int numGroups = colTiles / NT;
    const int rowTiles  = Nrows >> 4;
    if (waveId >= rowTiles * numGroups) return;     // whole wave exits together
    const int rowTile = waveId / numGroups;
    const int group   = waveId - rowTile * numGroups;
    const int rowBase = rowTile << 4;

    const int half = lane >> 4;     // 0: lanes 0-15, 1: lanes 16-31
    const int m    = lane & 15;     // A-row within tile; B/C column within tile
    const int kA   = half * 16;     // A fragment K offset (contiguous, matches packed B)
    const int Ksteps = K >> 5;

    const v16h* __restrict__ Bfrag = (const v16h*)Wpk;

    v8f acc[NT];
#pragma unroll
    for (int t = 0; t < NT; ++t) acc[t] = (v8f){};

    const float* arowBase = A + (size_t)(rowBase + m) * K;
    for (int kt = 0; kt < Ksteps; ++kt) {
        const float* arow = arowBase + kt * 32;
        v16h a;
#pragma unroll
        for (int j = 0; j < 8; ++j) a[j]     = (_Float16)arow[kA + j];
#pragma unroll
        for (int j = 0; j < 8; ++j) a[8 + j] = (_Float16)arow[kA + 8 + j];
#pragma unroll
        for (int t = 0; t < NT; ++t) {
            int tile = group * NT + t;
            v16h b = Bfrag[((size_t)tile * Ksteps + kt) * 32 + lane];
            acc[t] = __builtin_amdgcn_wmma_f32_16x16x32_f16(false, a, false, b,
                                                            (short)0, acc[t], false, false);
        }
    }

#pragma unroll
    for (int t = 0; t < NT; ++t) {
        int col = (group * NT + t) * 16 + m;
        if (col < Ncols) {
            float bv = bias[col];
#pragma unroll
            for (int i = 0; i < 8; ++i) {
                float v = acc[t][i] + bv;
                if (doRelu) v = fmaxf(v, 0.f);
                C[(size_t)(rowBase + i + 8 * half) * Ncols + col] = v;
            }
        }
    }
}

// ---------------------------------------------------------------------------
// Attention prep: vrh[k,h] = sum_c Wr[k, h*HID+c]*ar[h,c]; crh[h] = br.ar
// (lets alpha_r = x @ vrh + crh without materializing x@Wr)
// ---------------------------------------------------------------------------
__global__ void prep_vr(const float* __restrict__ Wr, const float* __restrict__ br,
                        const float* __restrict__ ar, float* __restrict__ vrh,
                        float* __restrict__ crh, int K)
{
    int i = blockIdx.x * blockDim.x + threadIdx.x;
    if (i < K * HEADS) {
        int k = i >> 1, h = i & 1;
        float s = 0.f;
        for (int c = 0; c < HID; ++c) s += Wr[(size_t)k * F + h * HID + c] * ar[h * HID + c];
        vrh[i] = s;
    } else if (i < K * HEADS + HEADS) {
        int h = i - K * HEADS;
        float s = 0.f;
        for (int c = 0; c < HID; ++c) s += br[h * HID + c] * ar[h * HID + c];
        crh[h] = s;
    }
}

__global__ void alpha_left(const float* __restrict__ xl, const float* __restrict__ al,
                           float* __restrict__ alpha, int Nn)
{
    int i = blockIdx.x * blockDim.x + threadIdx.x;
    if (i >= Nn * HEADS) return;
    int nid = i >> 1, h = i & 1;
    const float* row = xl + (size_t)nid * F + h * HID;
    float s = 0.f;
    for (int c = 0; c < HID; ++c) s += row[c] * al[h * HID + c];
    alpha[i] = s;
}

__global__ void alpha_right(const float* __restrict__ X, const float* __restrict__ vrh,
                            const float* __restrict__ crh, float* __restrict__ alpha,
                            int Nn, int K)
{
    int i = blockIdx.x * blockDim.x + threadIdx.x;
    if (i >= Nn * HEADS) return;
    int nid = i >> 1, h = i & 1;
    const float* row = X + (size_t)nid * K;
    float s = crh[h];
    for (int k = 0; k < K; ++k) s += row[k] * vrh[k * 2 + h];
    alpha[i] = s;
}

// ---------------------------------------------------------------------------
// Monotone uint encoding of float for atomicMax-based segment max
// ---------------------------------------------------------------------------
__device__ __forceinline__ unsigned fenc(float f) {
    unsigned b = __float_as_uint(f);
    return (b & 0x80000000u) ? ~b : (b | 0x80000000u);
}
__device__ __forceinline__ float fdec(unsigned k) {
    unsigned b = (k & 0x80000000u) ? (k ^ 0x80000000u) : ~k;
    return __uint_as_float(b);
}
__device__ __forceinline__ float lrelu(float v) { return v > 0.f ? v : NEG_SLOPE * v; }

__global__ void edge_max(const int* __restrict__ src, const int* __restrict__ dst,
                         const float* __restrict__ al, const float* __restrict__ ar,
                         unsigned* __restrict__ mkey, int E)
{
    int e = blockIdx.x * blockDim.x + threadIdx.x;
    if (e >= E) return;
    int s = src[e], d = dst[e];
#pragma unroll
    for (int h = 0; h < HEADS; ++h) {
        float v = lrelu(al[s * 2 + h] + ar[d * 2 + h]);
        atomicMax(&mkey[d * 2 + h], fenc(v));
    }
}

__global__ void edge_exp(const int* __restrict__ src, const int* __restrict__ dst,
                         const float* __restrict__ al, const float* __restrict__ ar,
                         const unsigned* __restrict__ mkey, float* __restrict__ ssum,
                         float* __restrict__ exbuf, int E)
{
    int e = blockIdx.x * blockDim.x + threadIdx.x;
    if (e >= E) return;
    int s = src[e], d = dst[e];
#pragma unroll
    for (int h = 0; h < HEADS; ++h) {
        float v = lrelu(al[s * 2 + h] + ar[d * 2 + h]);
        float t = expf(v - fdec(mkey[d * 2 + h]));
        exbuf[(size_t)e * 2 + h] = t;
        atomicAdd(&ssum[d * 2 + h], t);
    }
}

// one wave per edge: lane loads float4 of xl[src], 4 atomic adds to out[dst]
__global__ void edge_scatter(const int* __restrict__ src, const int* __restrict__ dst,
                             const float* __restrict__ xl, const float* __restrict__ ssum,
                             const float* __restrict__ exbuf, float* __restrict__ out, int E)
{
    int wavesPerBlock = blockDim.x >> 5;
    int e = blockIdx.x * wavesPerBlock + (threadIdx.x >> 5);
    if (e >= E) return;
    int lane = threadIdx.x & 31;
    int s = src[e], d = dst[e];
    int h = lane >> 4;                          // channels lane*4..lane*4+3 share a head
    float a = exbuf[(size_t)e * 2 + h] / (ssum[d * 2 + h] + 1e-16f);
    const float4 xv = *(const float4*)(xl + (size_t)s * F + lane * 4);
    float* o = out + (size_t)d * F + lane * 4;
    atomicAdd(o + 0, xv.x * a);
    atomicAdd(o + 1, xv.y * a);
    atomicAdd(o + 2, xv.z * a);
    atomicAdd(o + 3, xv.w * a);
}

__global__ void fill_zero(unsigned* __restrict__ p, size_t n)
{
    size_t i = (size_t)blockIdx.x * blockDim.x + threadIdx.x;
    if (i < n) p[i] = 0u;
}

__global__ void relu_ip(float* __restrict__ p, size_t n)
{
    size_t i = (size_t)blockIdx.x * blockDim.x + threadIdx.x;
    if (i < n) p[i] = fmaxf(p[i], 0.f);
}

__global__ void logsoftmax40(const float* __restrict__ logits, float* __restrict__ out, int Nn)
{
    int n = blockIdx.x * blockDim.x + threadIdx.x;
    if (n >= Nn) return;
    const float* r = logits + (size_t)n * 40;
    float mx = -INFINITY;
    for (int j = 0; j < 40; ++j) mx = fmaxf(mx, r[j]);
    float s = 0.f;
    for (int j = 0; j < 40; ++j) s += expf(r[j] - mx);
    float lse = mx + logf(s);
    for (int j = 0; j < 40; ++j) out[(size_t)n * 40 + j] = r[j] - lse;
}

// ---------------------------------------------------------------------------
// Host-side helpers
// ---------------------------------------------------------------------------
template<int NT>
static void launch_gemm(const float* A, const float* W, _Float16* Wpk,
                        const float* bias, float* C,
                        int Nn, int K, int Ncols, int doRelu, hipStream_t stream)
{
    int colTiles = (Ncols + 15) >> 4;
    int Ksteps   = K >> 5;
    int packN    = colTiles * Ksteps * 512;
    pack_w<<<(packN + 255) / 256, 256, 0, stream>>>(W, Wpk, K, Ncols, colTiles);
    int waves  = (Nn >> 4) * (colTiles / NT);
    int blocks = (waves + 7) / 8;
    gemm_wmma_packed<NT><<<blocks, 256, 0, stream>>>(A, Wpk, bias, C, Nn, K, Ncols, doRelu);
}

static void run_gat_layer(const float* X, int K,
                          const float* Wl, const float* bl,
                          const float* Wr, const float* br,
                          const float* al, const float* ar,
                          const int* srcI, const int* dstI, int Nn, int E,
                          float* xl, _Float16* Wpk, float* alpL, float* alpR,
                          unsigned* mkey, float* ssum, float* exbuf,
                          float* vrh, float* crh, float* outH, hipStream_t stream)
{
    // xl = X @ Wl + bl  (WMMA, packed B, 16x64 strip per wave)
    launch_gemm<4>(X, Wl, Wpk, bl, xl, Nn, K, F, 0, stream);

    prep_vr<<<(K * HEADS + HEADS + 255) / 256, 256, 0, stream>>>(Wr, br, ar, vrh, crh, K);
    alpha_left <<<(Nn * HEADS + 255) / 256, 256, 0, stream>>>(xl, al, alpL, Nn);
    alpha_right<<<(Nn * HEADS + 255) / 256, 256, 0, stream>>>(X, vrh, crh, alpR, Nn, K);

    size_t nf = (size_t)Nn * F;
    fill_zero<<<(unsigned)((nf + 255) / 256), 256, 0, stream>>>((unsigned*)outH, nf);
    fill_zero<<<(Nn * HEADS + 255) / 256, 256, 0, stream>>>(mkey, (size_t)Nn * HEADS);
    fill_zero<<<(Nn * HEADS + 255) / 256, 256, 0, stream>>>((unsigned*)ssum, (size_t)Nn * HEADS);

    edge_max    <<<(E + 255) / 256, 256, 0, stream>>>(srcI, dstI, alpL, alpR, mkey, E);
    edge_exp    <<<(E + 255) / 256, 256, 0, stream>>>(srcI, dstI, alpL, alpR, mkey, ssum, exbuf, E);
    edge_scatter<<<(E + 7) / 8,     256, 0, stream>>>(srcI, dstI, xl, ssum, exbuf, outH, E);

    relu_ip<<<(unsigned)((nf + 255) / 256), 256, 0, stream>>>(outH, nf);
}

extern "C" void kernel_launch(void* const* d_in, const int* in_sizes, int n_in,
                              void* d_out, int out_size, void* d_ws, size_t ws_size,
                              hipStream_t stream)
{
    const float* x   = (const float*)d_in[0];
    const int*   ei  = (const int*)  d_in[1];
    const float* Wl0 = (const float*)d_in[2];
    const float* bl0 = (const float*)d_in[3];
    const float* Wr0 = (const float*)d_in[4];
    const float* br0 = (const float*)d_in[5];
    const float* al0 = (const float*)d_in[6];
    const float* ar0 = (const float*)d_in[7];
    const float* Wl1 = (const float*)d_in[8];
    const float* bl1 = (const float*)d_in[9];
    const float* Wr1 = (const float*)d_in[10];
    const float* br1 = (const float*)d_in[11];
    const float* al1 = (const float*)d_in[12];
    const float* ar1 = (const float*)d_in[13];
    const float* Wm1 = (const float*)d_in[14];
    const float* bm1 = (const float*)d_in[15];
    const float* Wm2 = (const float*)d_in[16];
    const float* bm2 = (const float*)d_in[17];

    const int Nn = in_sizes[0] / 128;   // 50000
    const int E  = in_sizes[1] / 2;     // 1,600,000
    const int IN = 128;

    float* ws = (float*)d_ws;
    size_t off = 0;
    float*    xl   = ws + off; off += (size_t)Nn * F;
    float*    bufB = ws + off; off += (size_t)Nn * F;       // h0
    float*    bufC = ws + off; off += (size_t)Nn * F;       // h1
    float*    alpL = ws + off; off += (size_t)Nn * HEADS;
    float*    alpR = ws + off; off += (size_t)Nn * HEADS;
    unsigned* mkey = (unsigned*)(ws + off); off += (size_t)Nn * HEADS;
    float*    ssum = ws + off; off += (size_t)Nn * HEADS;
    float*    exbf = ws + off; off += (size_t)E * HEADS;
    _Float16* Wpk  = (_Float16*)(ws + off); off += 16384;   // packed B fragments (32B-aligned)
    float*    vrh  = ws + off; off += F * HEADS;
    float*    crh  = ws + off; off += HEADS;

    const int* srcI = ei;
    const int* dstI = ei + E;

    // ---- Layer 0 (input x, K=128) -> bufB ----
    run_gat_layer(x, IN, Wl0, bl0, Wr0, br0, al0, ar0, srcI, dstI, Nn, E,
                  xl, Wpk, alpL, alpR, mkey, ssum, exbf, vrh, crh, bufB, stream);
    // ---- Layer 1 (input bufB, K=128) -> bufC ----
    run_gat_layer(bufB, F, Wl1, bl1, Wr1, br1, al1, ar1, srcI, dstI, Nn, E,
                  xl, Wpk, alpL, alpR, mkey, ssum, exbf, vrh, crh, bufC, stream);

    // ---- post_mp: (h1 @ Wm1 + bm1) @ Wm2 + bm2, then log_softmax ----
    float* tmid   = xl;                     // reuse: N x 64
    float* logits = xl + (size_t)Nn * 64;   // N x 40 (fits inside xl region)
    launch_gemm<4>(bufC, Wm1, Wpk, bm1, tmid,  Nn, F,  64, 0, stream);   // 4 tiles, 1 group
    launch_gemm<3>(tmid, Wm2, Wpk, bm2, logits, Nn, 64, 40, 0, stream); // 3 tiles (padded)
    logsoftmax40<<<(Nn + 255) / 256, 256, 0, stream>>>(logits, (float*)d_out, Nn);
}
